// TFAttention_50242527428965
// MI455X (gfx1250) — compile-verified, matrix operand order checked by reading, not run on hardware
//
#include <hip/hip_runtime.h>

typedef __bf16 bf16;
typedef __attribute__((ext_vector_type(16))) __bf16 bf16x16;
typedef __attribute__((ext_vector_type(8)))  __bf16 bf16x8;
typedef __attribute__((ext_vector_type(8)))  float  f32x8;

#define B_  4
#define S_  2048
#define D_  1024
#define H_  16
#define HD_ 64

// ---- WMMA helpers (CDNA5 16x16x32 bf16, wave32) ----------------------------
// A-frag: lane m = lane&15 holds row m; element pairs map to
//   K = hi*8 + e (e<8) and K = 16 + hi*8 + (e-8)  with hi = lane>>4.
// => two contiguous 16B chunks per lane from a row-major (K-contiguous) row.
// B-frag uses the symmetric layout over a column-contiguous (n-major) store.
__device__ __forceinline__ bf16x16 load_frag(const bf16* rowptr, int hi) {
  union { bf16x16 v; bf16x8 h[2]; } f;
  f.h[0] = *(const bf16x8*)(rowptr + hi * 8);
  f.h[1] = *(const bf16x8*)(rowptr + 16 + hi * 8);
  return f.v;
}

__device__ __forceinline__ f32x8 wmma_bf16(bf16x16 a, bf16x16 b, f32x8 c) {
  return __builtin_amdgcn_wmma_f32_16x16x32_bf16(false, a, false, b,
                                                 (short)0, c, false, false);
}

// ---- CDNA5 async global->LDS staging (ASYNCcnt-tracked) --------------------
__device__ __forceinline__ void async_copy_b128(const bf16* gsrc, bf16* ldst) {
  unsigned lds = (unsigned)(uintptr_t)ldst;     // LDS aperture: low 32 = offset
  unsigned long long ga = (unsigned long long)(uintptr_t)gsrc;
  asm volatile("global_load_async_to_lds_b128 %0, %1, off"
               :: "v"(lds), "v"(ga) : "memory");
}
__device__ __forceinline__ void wait_async_zero() {
  asm volatile("s_wait_asynccnt 0x0" ::: "memory");
}

// ---- fp32 -> bf16 cast ------------------------------------------------------
__global__ __launch_bounds__(256) void cast_f32_to_bf16(
    const float* __restrict__ in, bf16* __restrict__ out, int n) {
  int i = (blockIdx.x * blockDim.x + threadIdx.x) * 4;
  if (i + 3 < n) {
    float4 v = *(const float4*)(in + i);
    out[i + 0] = (bf16)v.x; out[i + 1] = (bf16)v.y;
    out[i + 2] = (bf16)v.z; out[i + 3] = (bf16)v.w;
  } else {
    for (; i < n; ++i) out[i] = (bf16)in[i];
  }
}

// ---- fp32 [R][C] -> bf16 transposed [C][R] (LDS-tiled, one-time) -----------
__global__ __launch_bounds__(256) void cast_transpose_f32_bf16(
    const float* __restrict__ in, bf16* __restrict__ out,
    unsigned R, unsigned C) {
  __shared__ float tile[32][33];
  const unsigned bx = blockIdx.x * 32;  // col base
  const unsigned by = blockIdx.y * 32;  // row base
  const unsigned tx = threadIdx.x & 31, ty = threadIdx.x >> 5;  // 32x8
#pragma unroll
  for (int i = 0; i < 4; ++i)
    tile[ty + i * 8][tx] = in[(size_t)(by + ty + i * 8) * C + bx + tx];
  __syncthreads();
#pragma unroll
  for (int i = 0; i < 4; ++i)
    out[(size_t)(bx + ty + i * 8) * R + by + tx] = (bf16)tile[tx][ty + i * 8];
}

// ---- double-buffered GEMM mainloop: C[128x256] += A[128xK] * Wt[256xK]^T ---
// A row-major [M][K] bf16, Wt row-major [N][K] bf16 (pre-transposed weights).
// 8 waves in 2x4 grid; wave computes 64x64 via 4x4 wmma tiles (16 wmma/step).
// Async copies for stage i+1 overlap WMMA of stage i; one barrier per step.
__device__ __forceinline__ void stage_tiles(
    const bf16* __restrict__ A, const bf16* __restrict__ Wt,
    int K, int rowBase, int colBase, int k0,
    bf16* sA, bf16* sB, int tid) {
#pragma unroll
  for (int t = 0; t < 2; ++t) {             // A tile 128x32
    int v = tid + t * 256;
    int row = v >> 2, kv = (v & 3) * 8;
    async_copy_b128(A + (size_t)(rowBase + row) * K + k0 + kv,
                    sA + row * 32 + kv);
  }
#pragma unroll
  for (int t = 0; t < 4; ++t) {             // Wt tile 256x32
    int v = tid + t * 256;
    int row = v >> 2, kv = (v & 3) * 8;
    async_copy_b128(Wt + (size_t)(colBase + row) * K + k0 + kv,
                    sB + row * 32 + kv);
  }
}

__device__ __forceinline__ void gemm_mainloop(
    const bf16* __restrict__ A, const bf16* __restrict__ Wt,
    int K, int rowBase, int colBase,
    bf16* sA, bf16* sB, f32x8 acc[4][4]) {
  const int tid  = threadIdx.x;
  const int lane = tid & 31, wid = tid >> 5;
  const int hi = lane >> 4, ln = lane & 15;
  const int wr = wid >> 2,  wc = wid & 3;
  const int nsteps = K / 32;

  stage_tiles(A, Wt, K, rowBase, colBase, 0, sA, sB, tid);
  wait_async_zero();
  __syncthreads();

  for (int i = 0; i < nsteps; ++i) {
    const int cur = i & 1, nxt = cur ^ 1;
    if (i + 1 < nsteps)  // prefetch next stage into the other buffer
      stage_tiles(A, Wt, K, rowBase, colBase, (i + 1) * 32,
                  sA + nxt * (128 * 32), sB + nxt * (256 * 32), tid);

    const bf16* cA = sA + cur * (128 * 32);
    const bf16* cB = sB + cur * (256 * 32);
    bf16x16 af[4], bf_[4];
#pragma unroll
    for (int ii = 0; ii < 4; ++ii)
      af[ii] = load_frag(cA + (wr * 64 + ii * 16 + ln) * 32, hi);
#pragma unroll
    for (int j = 0; j < 4; ++j)
      bf_[j] = load_frag(cB + (wc * 64 + j * 16 + ln) * 32, hi);
#pragma unroll
    for (int ii = 0; ii < 4; ++ii)
#pragma unroll
      for (int j = 0; j < 4; ++j)
        acc[ii][j] = wmma_bf16(af[ii], bf_[j], acc[ii][j]);

    wait_async_zero();   // own prefetch copies done
    __syncthreads();     // everyone done computing cur + waiting prefetch
  }
}

// ---- QKV projection: qkv = x @ Wqkv + b, scattered into head layouts -------
// q,k -> [B,H,S,HD] bf16 ; v -> transposed [B,H,HD,S] bf16 (for P@V B-frags)
__global__ __launch_bounds__(256) void gemm_qkv_kernel(
    const bf16* __restrict__ X, const bf16* __restrict__ Wt,
    const float* __restrict__ bias,
    bf16* __restrict__ qb, bf16* __restrict__ kb, bf16* __restrict__ vtb) {
  __shared__ __align__(16) bf16 sA[2 * 128 * 32];
  __shared__ __align__(16) bf16 sB[2 * 256 * 32];
  f32x8 acc[4][4];
  const f32x8 z = {0, 0, 0, 0, 0, 0, 0, 0};
#pragma unroll
  for (int i = 0; i < 4; ++i)
#pragma unroll
    for (int j = 0; j < 4; ++j) acc[i][j] = z;

  const int rowBase = blockIdx.y * 128, colBase = blockIdx.x * 256;
  gemm_mainloop(X, Wt, D_, rowBase, colBase, sA, sB, acc);

  const unsigned lane = threadIdx.x & 31, wid = threadIdx.x >> 5;
  const unsigned hi = lane >> 4, ln = lane & 15;
  const unsigned wr = wid >> 2, wc = wid & 3;
#pragma unroll
  for (int i = 0; i < 4; ++i)
#pragma unroll
    for (int j = 0; j < 4; ++j)
#pragma unroll
      for (int v = 0; v < 8; ++v) {
        unsigned r = rowBase + wr * 64 + i * 16 + hi * 8 + v;
        unsigned c = colBase + wc * 64 + j * 16 + ln;
        bf16 bv = (bf16)(acc[i][j][v] + bias[c]);
        unsigned which = c >> 10, d = c & 1023;
        unsigned h = d >> 6, hd = d & 63;
        unsigned b = r >> 11, s = r & 2047;
        size_t head = (size_t)(b * H_ + h);
        if (which == 0)      qb [(head * S_ + s) * HD_ + hd] = bv;
        else if (which == 1) kb [(head * S_ + s) * HD_ + hd] = bv;
        else                 vtb[(head * HD_ + hd) * S_ + s] = bv;
      }
}

// ---- Flash attention: per block = (b,h,128 q-rows); wave = 16 q-rows -------
__global__ __launch_bounds__(256) void attn_kernel(
    const bf16* __restrict__ qb, const bf16* __restrict__ kb,
    const bf16* __restrict__ vtb, bf16* __restrict__ ab) {
  __shared__ __align__(16) bf16 pscr[8 * 16 * 32];  // per-wave P scratch
  const unsigned tid = threadIdx.x;
  const unsigned lane = tid & 31, w = tid >> 5;
  const unsigned hi = lane >> 4, ln = lane & 15;
  const unsigned qt = blockIdx.x & 15;
  const unsigned h  = (blockIdx.x >> 4) & 15;
  const unsigned b  = blockIdx.x >> 8;
  const unsigned q0 = qt * 128 + w * 16;
  const size_t head = (size_t)(b * H_ + h);

  const bf16* Q = qb + (head * S_ + q0) * HD_;
  bf16x16 qf0 = load_frag(Q + ln * HD_, hi);        // hd 0..31
  bf16x16 qf1 = load_frag(Q + ln * HD_ + 32, hi);   // hd 32..63

  const f32x8 z = {0, 0, 0, 0, 0, 0, 0, 0};
  f32x8 o[4] = {z, z, z, z};
  float mi[8], li[8];
#pragma unroll
  for (int v = 0; v < 8; ++v) { mi[v] = -3.0e38f; li[v] = 0.0f; }
  bf16* scr = pscr + w * (16 * 32);

  const unsigned ntiles = (q0 + 15) / 32 + 1;  // causal: keys <= q0+15
  for (unsigned kt = 0; kt < ntiles; ++kt) {
    const unsigned kbase = kt * 32;
    // S = Q @ K^T for 16q x 32keys (two 16x16 C-frags)
    f32x8 s0 = z, s1 = z;
    {
      const bf16* Kr0 = kb + (head * S_ + kbase + ln) * HD_;
      s0 = wmma_bf16(qf0, load_frag(Kr0, hi), s0);
      s0 = wmma_bf16(qf1, load_frag(Kr0 + 32, hi), s0);
      const bf16* Kr1 = kb + (head * S_ + kbase + 16 + ln) * HD_;
      s1 = wmma_bf16(qf0, load_frag(Kr1, hi), s1);
      s1 = wmma_bf16(qf1, load_frag(Kr1 + 32, hi), s1);
    }
    // scale + causal mask (reference semantics: masked = -10000 exactly)
    float rmax[8], rsum[8];
#pragma unroll
    for (int v = 0; v < 8; ++v) {
      unsigned qrow = q0 + hi * 8 + v;
      float a0 = s0[v] * 0.125f;            // 1/sqrt(64)
      float a1 = s1[v] * 0.125f;
      if (kbase + ln > qrow)      a0 = -10000.0f;
      if (kbase + 16 + ln > qrow) a1 = -10000.0f;
      s0[v] = a0; s1[v] = a1;
      rmax[v] = fmaxf(a0, a1);
    }
    // row reductions across the 16-lane half holding this row
#pragma unroll
    for (int msk = 1; msk < 16; msk <<= 1)
#pragma unroll
      for (int v = 0; v < 8; ++v)
        rmax[v] = fmaxf(rmax[v], __shfl_xor(rmax[v], msk, 32));
    float alpha[8];
#pragma unroll
    for (int v = 0; v < 8; ++v) {
      float mn = fmaxf(mi[v], rmax[v]);
      alpha[v] = __expf(mi[v] - mn);
      mi[v] = mn;
      float p0 = __expf(s0[v] - mn);
      float p1 = __expf(s1[v] - mn);
      rsum[v] = p0 + p1;
      // spill P (C-layout) to LDS as bf16 for A-frag reload
      scr[(hi * 8 + v) * 32 + ln]      = (bf16)p0;
      scr[(hi * 8 + v) * 32 + 16 + ln] = (bf16)p1;
    }
#pragma unroll
    for (int msk = 1; msk < 16; msk <<= 1)
#pragma unroll
      for (int v = 0; v < 8; ++v)
        rsum[v] += __shfl_xor(rsum[v], msk, 32);
#pragma unroll
    for (int v = 0; v < 8; ++v) li[v] = li[v] * alpha[v] + rsum[v];
#pragma unroll
    for (int j = 0; j < 4; ++j)
#pragma unroll
      for (int v = 0; v < 8; ++v) o[j][v] *= alpha[v];

    asm volatile("s_wait_dscnt 0x0" ::: "memory");  // P visible across lanes
    bf16x16 pf = load_frag(scr + ln * 32, hi);      // A-frag of P (K=32 keys)
#pragma unroll
    for (int j = 0; j < 4; ++j) {
      const bf16* Vc = vtb + (head * HD_ + j * 16 + ln) * S_ + kbase;
      o[j] = wmma_bf16(pf, load_frag(Vc, hi), o[j]);
    }
  }
  // normalize and emit a = softmax(S)V in [B,S,D] bf16
#pragma unroll
  for (int v = 0; v < 8; ++v) li[v] = 1.0f / li[v];
#pragma unroll
  for (int j = 0; j < 4; ++j)
#pragma unroll
    for (int v = 0; v < 8; ++v) {
      unsigned row = q0 + hi * 8 + v;
      unsigned col = h * HD_ + j * 16 + ln;
      ab[((size_t)b * S_ + row) * D_ + col] = (bf16)(o[j][v] * li[v]);
    }
}

// ---- output projection: out = a @ Wproj + b (fp32 out) ---------------------
__global__ __launch_bounds__(256) void gemm_proj_kernel(
    const bf16* __restrict__ A, const bf16* __restrict__ Wt,
    const float* __restrict__ bias, float* __restrict__ out) {
  __shared__ __align__(16) bf16 sA[2 * 128 * 32];
  __shared__ __align__(16) bf16 sB[2 * 256 * 32];
  f32x8 acc[4][4];
  const f32x8 z = {0, 0, 0, 0, 0, 0, 0, 0};
#pragma unroll
  for (int i = 0; i < 4; ++i)
#pragma unroll
    for (int j = 0; j < 4; ++j) acc[i][j] = z;

  const int rowBase = blockIdx.y * 128, colBase = blockIdx.x * 256;
  gemm_mainloop(A, Wt, D_, rowBase, colBase, sA, sB, acc);

  const unsigned lane = threadIdx.x & 31, wid = threadIdx.x >> 5;
  const unsigned hi = lane >> 4, ln = lane & 15;
  const unsigned wr = wid >> 2, wc = wid & 3;
#pragma unroll
  for (int i = 0; i < 4; ++i)
#pragma unroll
    for (int j = 0; j < 4; ++j)
#pragma unroll
      for (int v = 0; v < 8; ++v) {
        unsigned r = rowBase + wr * 64 + i * 16 + hi * 8 + v;
        unsigned c = colBase + wc * 64 + j * 16 + ln;
        out[(size_t)r * D_ + c] = acc[i][j][v] + bias[c];
      }
}

// ---- host-side orchestration ----------------------------------------------
extern "C" void kernel_launch(void* const* d_in, const int* in_sizes, int n_in,
                              void* d_out, int out_size, void* d_ws,
                              size_t ws_size, hipStream_t stream) {
  const float* x        = (const float*)d_in[0];
  const float* c_attn_w = (const float*)d_in[1];
  const float* c_attn_b = (const float*)d_in[2];
  const float* c_proj_w = (const float*)d_in[3];
  const float* c_proj_b = (const float*)d_in[4];
  float* out = (float*)d_out;

  char* ws = (char*)d_ws;
  size_t off = 0;
  auto alloc = [&](size_t bytes) -> void* {
    void* p = ws + off;
    off += (bytes + 255) & ~(size_t)255;
    return p;
  };
  bf16* xb    = (bf16*)alloc((size_t)B_ * S_ * D_ * 2);
  bf16* wqkvT = (bf16*)alloc((size_t)3 * D_ * D_ * 2);   // [3D][D]
  bf16* wprojT= (bf16*)alloc((size_t)D_ * D_ * 2);       // [D][D]
  bf16* qb    = (bf16*)alloc((size_t)B_ * H_ * S_ * HD_ * 2);
  bf16* kb    = (bf16*)alloc((size_t)B_ * H_ * S_ * HD_ * 2);
  bf16* vtb   = (bf16*)alloc((size_t)B_ * H_ * S_ * HD_ * 2);
  bf16* ab    = (bf16*)alloc((size_t)B_ * S_ * D_ * 2);

  const int nx = B_ * S_ * D_;
  cast_f32_to_bf16<<<(nx / 4 + 255) / 256, 256, 0, stream>>>(x, xb, nx);
  // transpose-cast weights: Wt[n][k] = W[k][n]
  cast_transpose_f32_bf16<<<dim3(3 * D_ / 32, D_ / 32), 256, 0, stream>>>(
      c_attn_w, wqkvT, D_, 3 * D_);
  cast_transpose_f32_bf16<<<dim3(D_ / 32, D_ / 32), 256, 0, stream>>>(
      c_proj_w, wprojT, D_, D_);

  dim3 g1((3 * D_) / 256, (B_ * S_) / 128);
  gemm_qkv_kernel<<<g1, 256, 0, stream>>>(xb, wqkvT, c_attn_b, qb, kb, vtb);

  attn_kernel<<<B_ * H_ * (S_ / 128), 256, 0, stream>>>(qb, kb, vtb, ab);

  dim3 g2(D_ / 256, (B_ * S_) / 128);
  gemm_proj_kernel<<<g2, 256, 0, stream>>>(ab, wprojT, c_proj_b, out);
}
